// DualAttentionLora_42726334661265
// MI455X (gfx1250) — matrix operand order repaired and checked by reading, compiled, not run on hardware
//
#include <hip/hip_runtime.h>
#include <hip/hip_bf16.h>

typedef __attribute__((ext_vector_type(16))) _Float16 v16h;
typedef __attribute__((ext_vector_type(8)))  _Float16 h8v;
typedef __attribute__((ext_vector_type(8)))  float    v8f;
typedef __attribute__((ext_vector_type(4)))  float    f4v;

#define NTOK 33049          // tokens per batch (M + 32768)
#define NB   4
#define MP   281            // "p" tokens
#define MPAD 288            // 281 padded to 18*16
#define NH   32768          // "h" tokens
#define CIN  256
#define DH   128
#define TTOT (NB * NTOK)    // 132196
#define QSCALE 0.125f       // 64^-0.5
#define LORA_SC 2.0f        // 16/8

// ---------------- WMMA helpers (layouts per CDNA5 ISA) ----------------
// A 16x32 f16 : lane L holds row M=L%16, K = (L/16)*16 + {0..15} (contiguous, same order as B)
// B 32x16 f16 : lane L holds col N=L%16, K = (L/16)*16 + {0..15}  (contiguous 32B; this code's order, not AMD's table order)
// C/D 16x16 f32: lane L holds col N=L%16, rows M = (L/16)*8 + vgpr

__device__ __forceinline__ v8f wmma16(v16h a, v16h b, v8f c) {
  return __builtin_amdgcn_wmma_f32_16x16x32_f16(false, a, false, b, (short)0, c, false, false);
}

// build A-operand from f16 memory: p points at (row, kbase + g*16)
__device__ __forceinline__ v16h load_a16(const _Float16* p) {
  h8v lo = *(const h8v*)p;
  h8v hi = *(const h8v*)(p + 8);
  v16h a;
#pragma unroll
  for (int i = 0; i < 8; i++) { a[i] = lo[i]; a[8 + i] = hi[i]; }
  return a;
}

// ---------------- K0: fold LoRA + scales into one f16 weight matrix ----------------
// Wcat rows: [0,128)=q (SCALE folded), [128,256)=k, [256,384)=v, [384,512)=res (res_w folded)
__global__ void prep_weights(const float* qW, const float* qb, const float* qA, const float* qB,
                             const float* kvW, const float* kvb, const float* kvA, const float* kvB,
                             const float* resW, const float* res_w,
                             _Float16* Wcat, float* biasCat) {
  int o = blockIdx.x;    // 0..511
  int i = threadIdx.x;   // 0..255
  float wv, bias = 0.f;
  if (o < 128) {
    float acc = qW[o * 256 + i];
#pragma unroll
    for (int r = 0; r < 8; r++) acc += LORA_SC * qA[o * 8 + r] * qB[r * 256 + i];
    wv = acc * QSCALE;
    bias = qb[o] * QSCALE;
  } else if (o < 384) {
    int oo = o - 128;
    float acc = kvW[oo * 256 + i];
#pragma unroll
    for (int r = 0; r < 8; r++) acc += LORA_SC * kvA[oo * 8 + r] * kvB[r * 256 + i];
    wv = acc;
    bias = kvb[oo];
  } else {
    wv = res_w[0] * resW[(o - 384) * 256 + i];
    bias = 0.f;
  }
  Wcat[o * 256 + i] = (_Float16)wv;
  if (i == 0) biasCat[o] = bias;
}

// ---------------- K1: fused projection GEMM  [T x 256] @ [256 x 512] ----------------
// One wave per block: 16 tokens x 128 outputs (otile = blockIdx.y selects q/k/v/res).
__global__ __launch_bounds__(32) void proj_kernel(const float* x, const _Float16* Wcat,
                                                  const float* biasCat,
                                                  _Float16* qbuf, _Float16* kbuf, _Float16* vbuf,
                                                  float* dout) {
  int lane = threadIdx.x, r = lane & 15, g = lane >> 4;
  int mtile = blockIdx.x, otile = blockIdx.y;
  int trow = mtile * 16 + r;
  if (trow >= TTOT) trow = TTOT - 1;  // clamp A reads on ragged tail

  v8f acc[8];
#pragma unroll
  for (int t = 0; t < 8; t++)
#pragma unroll
    for (int e = 0; e < 8; e++) acc[t][e] = 0.f;

  const float* px = x + (size_t)trow * CIN;
#pragma unroll
  for (int ks = 0; ks < 8; ks++) {
    int kb = ks * 32;
    // A: convert f32 x -> f16 on load
    const float* p = px + kb + g * 16;
    f4v x0 = *(const f4v*)p, x1 = *(const f4v*)(p + 4);
    f4v y0 = *(const f4v*)(p + 8), y1 = *(const f4v*)(p + 12);
    v16h a;
#pragma unroll
    for (int i = 0; i < 4; i++) {
      a[i] = (_Float16)x0[i]; a[4 + i] = (_Float16)x1[i];
      a[8 + i] = (_Float16)y0[i]; a[12 + i] = (_Float16)y1[i];
    }
#pragma unroll
    for (int nt = 0; nt < 8; nt++) {
      const _Float16* pw = Wcat + ((size_t)(otile * 128 + nt * 16 + r)) * CIN + kb + g * 16;
      acc[nt] = wmma16(a, *(const v16h*)pw, acc[nt]);
    }
  }
#pragma unroll
  for (int nt = 0; nt < 8; nt++) {
    int col = nt * 16 + r;  // column within this otile's 128
    float bc = biasCat[otile * 128 + col];
#pragma unroll
    for (int e = 0; e < 8; e++) {
      int tok = mtile * 16 + g * 8 + e;
      if (tok < TTOT) {
        float val = acc[nt][e] + bc;
        size_t idx = (size_t)tok * DH + col;
        if (otile == 0)      qbuf[idx] = (_Float16)val;
        else if (otile == 1) kbuf[idx] = (_Float16)val;
        else if (otile == 2) vbuf[idx] = (_Float16)val;
        else                 dout[idx] = val;  // residual path initializes d_out
      }
    }
  }
}

// ---------------- K2a: transpose v_p -> v_pT [b][128][288] (zero-padded) ----------------
__global__ void transpose_p(const _Float16* src, _Float16* dst) {
  int b = blockIdx.y;
  int idx = blockIdx.x * 256 + threadIdx.x;  // 0 .. 128*288-1
  if (idx >= DH * MPAD) return;
  int c = idx / MPAD, t = idx - c * MPAD;
  _Float16 val = (t < MP) ? src[((size_t)b * NTOK + t) * DH + c] : (_Float16)0.f;
  dst[((size_t)b * DH + c) * MPAD + t] = val;
}

// ---------------- K2b: transpose v_h -> v_hT [b][128][32768] ----------------
__global__ void transpose_h(const _Float16* src, _Float16* dst) {
  int b = blockIdx.y;
  int idx = blockIdx.x * 256 + threadIdx.x;  // 0 .. 128*32768-1
  int c = idx >> 15, t = idx & (NH - 1);
  dst[((size_t)b * DH + c) * NH + t] = src[((size_t)b * NTOK + MP + t) * DH + c];
}

// ---------------- K3: attn_h tile kernel (used twice) ----------------
// mode 0: out_h = softmax(q_h k_p^T) @ v_p  -> stored TRANSPOSED as out_hT [b][128][32768] f16
// mode 1: xh    = softmax(q_h k_p^T) @ out_p, B2 = out_pT;  d_out[h tokens] += xh
__global__ __launch_bounds__(32) void attn_h_kernel(const _Float16* qbuf, const _Float16* kbuf,
                                                    const _Float16* B2 /* [b][128][288] */,
                                                    _Float16* outT, float* dout, int mode) {
  __shared__ float    S[16][MPAD];
  __shared__ _Float16 P[16][MPAD];
  int lane = threadIdx.x, r = lane & 15, g = lane >> 4;
  int b = blockIdx.y, mtile = blockIdx.x;  // 0..2047
  size_t hbase = (size_t)b * NTOK + MP;

  // preload A (16 q_h rows x 128) as 4 K-steps
  v16h Aq[4];
  const _Float16* qp = qbuf + (hbase + mtile * 16 + r) * DH;
#pragma unroll
  for (int ks = 0; ks < 4; ks++) Aq[ks] = load_a16(qp + ks * 32 + g * 16);

  // S = q_h_tile @ k_p^T  (pad cols 281..287, masked in softmax)
#pragma unroll 2
  for (int nt = 0; nt < 18; nt++) {
    v8f acc;
#pragma unroll
    for (int e = 0; e < 8; e++) acc[e] = 0.f;
    const _Float16* kp = kbuf + ((size_t)b * NTOK + nt * 16 + r) * DH + g * 16;
#pragma unroll
    for (int ks = 0; ks < 4; ks++) acc = wmma16(Aq[ks], *(const v16h*)(kp + ks * 32), acc);
#pragma unroll
    for (int e = 0; e < 8; e++) S[g * 8 + e][nt * 16 + r] = acc[e];
  }
  __syncthreads();

  if (lane < 16) {  // exact softmax over 281 keys, normalized into P (f16)
    float m = -3.0e38f;
    for (int nn = 0; nn < MP; ++nn) m = fmaxf(m, S[lane][nn]);
    float s = 0.f;
    for (int nn = 0; nn < MP; ++nn) { float e = __expf(S[lane][nn] - m); S[lane][nn] = e; s += e; }
    float inv = 1.0f / s;
    for (int nn = 0; nn < MPAD; ++nn)
      P[lane][nn] = (nn < MP) ? (_Float16)(S[lane][nn] * inv) : (_Float16)0.f;
  }
  __syncthreads();

  // O = P(16x288) @ B2col(288x128)
#pragma unroll 2
  for (int nt2 = 0; nt2 < 8; nt2++) {
    v8f o;
#pragma unroll
    for (int e = 0; e < 8; e++) o[e] = 0.f;
    const _Float16* pb = B2 + ((size_t)b * DH + nt2 * 16 + r) * MPAD + g * 16;
#pragma unroll
    for (int ks = 0; ks < 9; ks++) {
      v16h a = load_a16(&P[r][ks * 32 + g * 16]);
      o = wmma16(a, *(const v16h*)(pb + ks * 32), o);
    }
    if (mode == 0) {  // store transposed out_h: lane's 8 rows are contiguous
      h8v hv;
#pragma unroll
      for (int e = 0; e < 8; e++) hv[e] = (_Float16)o[e];
      *(h8v*)(outT + ((size_t)b * DH + nt2 * 16 + r) * NH + mtile * 16 + g * 8) = hv;
    } else {
#pragma unroll
      for (int e = 0; e < 8; e++) {
        size_t di = (hbase + mtile * 16 + g * 8 + e) * DH + nt2 * 16 + r;
        dout[di] += o[e];
      }
    }
  }
}

// ---------------- K4: attn_p flash kernel ----------------
// Online softmax over 32768 keys; accumulates out_p = P@v_h AND xp = P@out_h together.
// Writes out_pT [b][128][288] (rows >=281 skipped) and d_out[p tokens] += xp.
__global__ __launch_bounds__(32) void attn_p_kernel(const _Float16* qbuf, const _Float16* kbuf,
                                                    const _Float16* vhT, const _Float16* ohT,
                                                    _Float16* outPT, float* dout) {
  __shared__ float    S[16][32];
  __shared__ _Float16 P[16][32];
  __shared__ float    mrun[16], lrun[16], alpha[16];
  int lane = threadIdx.x, r = lane & 15, g = lane >> 4;
  int b = blockIdx.y, mtile = blockIdx.x;  // 0..17

  v16h Aq[4];
  const _Float16* qp = qbuf + ((size_t)b * NTOK + mtile * 16 + r) * DH;
#pragma unroll
  for (int ks = 0; ks < 4; ks++) Aq[ks] = load_a16(qp + ks * 32 + g * 16);

  if (lane < 16) { mrun[lane] = -3.0e38f; lrun[lane] = 0.f; }
  v8f O[8], X[8];
#pragma unroll
  for (int t = 0; t < 8; t++)
#pragma unroll
    for (int e = 0; e < 8; e++) { O[t][e] = 0.f; X[t][e] = 0.f; }
  __syncthreads();

  const _Float16* kh = kbuf + ((size_t)b * NTOK + MP) * DH;
  for (int nc = 0; nc < NH / 32; ++nc) {
    int tok0 = nc * 32;
    if (nc + 1 < NH / 32)  // hide key-tile latency: CDNA5 global_prefetch_b8
      __builtin_prefetch(kh + (size_t)(tok0 + 32 + r) * DH, 0, 0);
    // S tile 16x32
#pragma unroll
    for (int st = 0; st < 2; st++) {
      v8f acc;
#pragma unroll
      for (int e = 0; e < 8; e++) acc[e] = 0.f;
      const _Float16* kp = kh + (size_t)(tok0 + st * 16 + r) * DH + g * 16;
#pragma unroll
      for (int ks = 0; ks < 4; ks++) acc = wmma16(Aq[ks], *(const v16h*)(kp + ks * 32), acc);
#pragma unroll
      for (int e = 0; e < 8; e++) S[g * 8 + e][st * 16 + r] = acc[e];
    }
    __syncthreads();
    if (lane < 16) {  // online softmax update for this 32-wide chunk
      float m0 = mrun[lane], cm = m0;
#pragma unroll 4
      for (int j = 0; j < 32; ++j) cm = fmaxf(cm, S[lane][j]);
      float al = __expf(m0 - cm), s = 0.f;
#pragma unroll 4
      for (int j = 0; j < 32; ++j) {
        float e = __expf(S[lane][j] - cm);
        P[lane][j] = (_Float16)e;
        s += e;
      }
      alpha[lane] = al; mrun[lane] = cm; lrun[lane] = lrun[lane] * al + s;
    }
    __syncthreads();
    float a8[8];
#pragma unroll
    for (int e = 0; e < 8; e++) a8[e] = alpha[g * 8 + e];
    v16h ap = load_a16(&P[r][g * 16]);  // P tile as A-operand (K=32)
#pragma unroll
    for (int nt2 = 0; nt2 < 8; nt2++) {
#pragma unroll
      for (int e = 0; e < 8; e++) { O[nt2][e] *= a8[e]; X[nt2][e] *= a8[e]; }
      const _Float16* pv = vhT + ((size_t)b * DH + nt2 * 16 + r) * NH + tok0 + g * 16;
      O[nt2] = wmma16(ap, *(const v16h*)pv, O[nt2]);
      const _Float16* po = ohT + ((size_t)b * DH + nt2 * 16 + r) * NH + tok0 + g * 16;
      X[nt2] = wmma16(ap, *(const v16h*)po, X[nt2]);
    }
    __syncthreads();
  }

  float il8[8];
#pragma unroll
  for (int e = 0; e < 8; e++) il8[e] = 1.0f / lrun[g * 8 + e];
#pragma unroll
  for (int nt2 = 0; nt2 < 8; nt2++) {
    int col = nt2 * 16 + r;
#pragma unroll
    for (int e = 0; e < 8; e++) {
      int mrow = mtile * 16 + g * 8 + e;
      if (mrow < MP) {
        outPT[((size_t)b * DH + col) * MPAD + mrow] = (_Float16)(O[nt2][e] * il8[e]);
        dout[((size_t)b * NTOK + mrow) * DH + col] += X[nt2][e] * il8[e];
      }
    }
  }
}

// ---------------- host launcher ----------------
extern "C" void kernel_launch(void* const* d_in, const int* in_sizes, int n_in,
                              void* d_out, int out_size, void* d_ws, size_t ws_size,
                              hipStream_t stream) {
  (void)in_sizes; (void)n_in; (void)out_size; (void)ws_size;
  const float* x    = (const float*)d_in[0];
  const float* qW   = (const float*)d_in[1];
  const float* qb   = (const float*)d_in[2];
  const float* qA   = (const float*)d_in[3];
  const float* qB   = (const float*)d_in[4];
  const float* kvW  = (const float*)d_in[5];
  const float* kvb  = (const float*)d_in[6];
  const float* kvA  = (const float*)d_in[7];
  const float* kvB  = (const float*)d_in[8];
  const float* resW = (const float*)d_in[9];
  const float* resw = (const float*)d_in[10];
  float* dout = (float*)d_out;

  char* w = (char*)d_ws;
  size_t off = 0;
  auto alloc = [&](size_t bytes) -> char* {
    char* p = w + off;
    off = (off + bytes + 255) & ~(size_t)255;
    return p;
  };
  _Float16* Wcat    = (_Float16*)alloc((size_t)512 * 256 * 2);
  float*    biasCat = (float*)   alloc((size_t)512 * 4);
  _Float16* qbuf    = (_Float16*)alloc((size_t)TTOT * DH * 2);
  _Float16* kbuf    = (_Float16*)alloc((size_t)TTOT * DH * 2);
  _Float16* vbuf    = (_Float16*)alloc((size_t)TTOT * DH * 2);
  _Float16* vpT     = (_Float16*)alloc((size_t)NB * DH * MPAD * 2);
  _Float16* vhT     = (_Float16*)alloc((size_t)NB * DH * NH * 2);
  _Float16* ohT     = (_Float16*)alloc((size_t)NB * DH * NH * 2);
  _Float16* opT     = (_Float16*)alloc((size_t)NB * DH * MPAD * 2);

  // 0) fold LoRA / SCALE / res_w into one f16 weight matrix
  prep_weights<<<dim3(512), dim3(256), 0, stream>>>(qW, qb, qA, qB, kvW, kvb, kvA, kvB,
                                                    resW, resw, Wcat, biasCat);
  // 1) fused projection GEMM (also writes residual part of d_out)
  proj_kernel<<<dim3((TTOT + 15) / 16, 4), dim3(32), 0, stream>>>(x, Wcat, biasCat,
                                                                  qbuf, kbuf, vbuf, dout);
  // 2) transposed copies for contiguous WMMA B-operands
  transpose_p<<<dim3((DH * MPAD + 255) / 256, NB), dim3(256), 0, stream>>>(vbuf, vpT);
  transpose_h<<<dim3((DH * NH) / 256, NB), dim3(256), 0, stream>>>(vbuf, vhT);
  // 3) out_h = softmax(q_h k_p^T) @ v_p   (stored transposed)
  attn_h_kernel<<<dim3(NH / 16, NB), dim3(32), 0, stream>>>(qbuf, kbuf, vpT, ohT, dout, 0);
  // 4) flash pass: out_p = P@v_h, xp = P@out_h; d_out[p] += xp
  attn_p_kernel<<<dim3(MPAD / 16, NB), dim3(32), 0, stream>>>(qbuf, kbuf, vhT, ohT, opT, dout);
  // 5) xh = softmax(q_h k_p^T) @ out_p; d_out[h] += xh
  attn_h_kernel<<<dim3(NH / 16, NB), dim3(32), 0, stream>>>(qbuf, kbuf, opT, ohT, dout, 1);
}
